// LSA_1726576853982
// MI455X (gfx1250) — matrix operand order repaired and checked by reading, not compiled, not run on hardware
//
#include <hip/hip_runtime.h>

typedef __attribute__((ext_vector_type(16))) _Float16 v16h;
typedef __attribute__((ext_vector_type(8)))  _Float16 v8h;
typedef __attribute__((ext_vector_type(8)))  float    v8f;
typedef __attribute__((ext_vector_type(4)))  int      v4i;

#define WMMA_F32_F16(a,b,c) \
  __builtin_amdgcn_wmma_f32_16x16x32_f16(false,(a),false,(b),(short)0,(c),false,false)

// ---- CDNA5 async global->LDS path (guarded so the file always compiles) ----
// Probe round 2 confirmed the builtin exists with prototype
//   void (int4 addrspace(1)*, int4 addrspace(3)*, imm int offset, imm int cpol)
#if defined(__gfx1250__) && __has_builtin(__builtin_amdgcn_global_load_async_to_lds_b128)
#define HAVE_ASYNC_LDS 1
#define ASYNC_COPY_B128(gp, lp)                                        \
  __builtin_amdgcn_global_load_async_to_lds_b128(                      \
      (__attribute__((address_space(1))) v4i*)(gp),                    \
      (__attribute__((address_space(3))) v4i*)(lp), 0, 0)
#if __has_builtin(__builtin_amdgcn_s_wait_asynccnt)
#define WAIT_ASYNC() __builtin_amdgcn_s_wait_asynccnt(0)
#else
#define WAIT_ASYNC() asm volatile("s_wait_asynccnt 0x0" ::: "memory")
#endif
#else
#define HAVE_ASYNC_LDS 0
#define WAIT_ASYNC() ((void)0)
#endif

static __device__ __forceinline__ v16h frag_cat(v8h lo, v8h hi){
  return __builtin_shufflevector(lo, hi, 0,1,2,3,4,5,6,7,8,9,10,11,12,13,14,15);
}
// A-fragment (16x32 f16): lane holds row (lane%16); chunks at k=(lane/16)*8 and +16
static __device__ __forceinline__ v16h load_frag_a(const _Float16* rowp, int lhi){
  v8h lo = *(const v8h*)(rowp + lhi*8);
  v8h hi = *(const v8h*)(rowp + lhi*8 + 16);
  return frag_cat(lo, hi);
}
// B-fragment (32x16 f16) from K-major row of B^T: lane holds col (lane%16);
// chunks at k=(lane/16)*8 and +16
static __device__ __forceinline__ v16h load_frag_b(const _Float16* rowp, int lhi){
  v8h lo = *(const v8h*)(rowp + lhi*8);
  v8h hi = *(const v8h*)(rowp + lhi*8 + 16);
  return frag_cat(lo, hi);
}

// ---------------------------------------------------------------------------
// Generic f16 WMMA GEMM: C[M,N] = A[M,K] * B[K,N]   (B supplied transposed)
// mode 0: Cout = acc + bias (f32)
// mode 1: scatter QKV columns into per-head [b,h,n,d] f16 buffers
// Block tile 128x128, K-step 32, 8 waves (4 along M x 2 along N).
// Double-buffered LDS; async global->LDS copies overlap the WMMAs.
// ---------------------------------------------------------------------------
__global__ __launch_bounds__(256) void gemm_wmma_f16(
    const _Float16* __restrict__ A, const _Float16* __restrict__ Bt,
    int Kdim, int Ncols, int mode,
    float* __restrict__ Cout, const float* __restrict__ bias,
    _Float16* __restrict__ Qd, _Float16* __restrict__ Kd, _Float16* __restrict__ Vd)
{
  __shared__ _Float16 sA[2][128*40];
  __shared__ _Float16 sB[2][128*40];
  const int tid  = threadIdx.x;
  const int lane = tid & 31;
  const int wid  = tid >> 5;
  const int wm   = wid & 3, wn = wid >> 2;
  const int lrow = lane & 15, lhi = lane >> 4;
  const long m0 = (long)blockIdx.x * 128;
  const long n0 = (long)blockIdx.y * 128;

  // this thread's copy slots: rows r and r+64, 8-half chunk at column c
  const int r = tid >> 2, c = (tid & 3)*8;

  const v8f z8 = {0.f,0.f,0.f,0.f,0.f,0.f,0.f,0.f};
  v8f acc[2][4];
  #pragma unroll
  for (int i=0;i<2;++i)
    #pragma unroll
    for (int j=0;j<4;++j) acc[i][j]=z8;

  auto copy_tile = [&](int buf, int kb){
    const _Float16* ga0 = A  + (m0 + r     )*Kdim + kb + c;
    const _Float16* ga1 = A  + (m0 + r + 64)*Kdim + kb + c;
    const _Float16* gb0 = Bt + (n0 + r     )*Kdim + kb + c;
    const _Float16* gb1 = Bt + (n0 + r + 64)*Kdim + kb + c;
    _Float16* la0 = &sA[buf][ r      *40 + c];
    _Float16* la1 = &sA[buf][(r + 64)*40 + c];
    _Float16* lb0 = &sB[buf][ r      *40 + c];
    _Float16* lb1 = &sB[buf][(r + 64)*40 + c];
#if HAVE_ASYNC_LDS
    ASYNC_COPY_B128(ga0, la0);
    ASYNC_COPY_B128(ga1, la1);
    ASYNC_COPY_B128(gb0, lb0);
    ASYNC_COPY_B128(gb1, lb1);
#else
    *(v8h*)la0 = *(const v8h*)ga0;
    *(v8h*)la1 = *(const v8h*)ga1;
    *(v8h*)lb0 = *(const v8h*)gb0;
    *(v8h*)lb1 = *(const v8h*)gb1;
#endif
  };

  copy_tile(0, 0);
  WAIT_ASYNC();
  __syncthreads();

  int cur = 0;
  for (int kb=0; kb<Kdim; kb+=32){
    const bool more = (kb + 32 < Kdim);
    const int nxt = cur ^ 1;
    if (more) copy_tile(nxt, kb + 32);          // overlaps with WMMAs below
    if (kb + 64 < Kdim){                        // L2 prefetch one slab further
      __builtin_prefetch(A  + (m0 + r)*Kdim + kb + 64, 0, 1);
      __builtin_prefetch(Bt + (n0 + r)*Kdim + kb + 64, 0, 1);
    }

    v16h af[2], bf[4];
    #pragma unroll
    for (int tm=0;tm<2;++tm)
      af[tm] = load_frag_a(&sA[cur][(wm*32 + tm*16 + lrow)*40], lhi);
    #pragma unroll
    for (int tn=0;tn<4;++tn)
      bf[tn] = load_frag_b(&sB[cur][(wn*64 + tn*16 + lrow)*40], lhi);
    #pragma unroll
    for (int tm=0;tm<2;++tm)
      #pragma unroll
      for (int tn=0;tn<4;++tn)
        acc[tm][tn] = WMMA_F32_F16(af[tm], bf[tn], acc[tm][tn]);

    if (more) WAIT_ASYNC();                     // my 4 copies for `nxt` done
    __syncthreads();                            // everyone's copies visible
    cur = nxt;
  }

  // C/D layout: lane -> col lane%16, rows (lane/16)*8 + i
  #pragma unroll
  for (int tm=0;tm<2;++tm){
    #pragma unroll
    for (int tn=0;tn<4;++tn){
      const long mbase = m0 + wm*32 + tm*16 + lhi*8;
      const long col   = n0 + wn*64 + tn*16 + lrow;
      if (mode == 0){
        float bv = bias[col];
        #pragma unroll
        for (int i=0;i<8;++i)
          Cout[(mbase + i)*Ncols + col] = acc[tm][tn][i] + bv;
      } else {
        const int sel = (int)(col >> 9);          // which of q/k/v
        const int c2  = (int)(col & 511);
        const int h   = c2 >> 6, d = c2 & 63;
        _Float16* dst = sel==0 ? Qd : (sel==1 ? Kd : Vd);
        #pragma unroll
        for (int i=0;i<8;++i){
          long m = mbase + i;
          long b = m >> 11;                        // / 2048
          long n = m & 2047;
          dst[(((b<<3) + h)*2048 + n)*64 + d] = (_Float16)acc[tm][tn][i];
        }
      }
    }
  }
}

// ---------------------------------------------------------------------------
// Flash attention with WMMA. One block = one (b,h) x 128 queries; each wave
// owns 16 query rows (Q fragments live in registers across the whole loop).
// Key/value blocks of 32; online softmax; diagonal masked to -FLT_MAX.
// K tile staged with async global->LDS; V transposed through VGPRs.
// ---------------------------------------------------------------------------
__global__ __launch_bounds__(256) void attn_wmma(
    const _Float16* __restrict__ Qd, const _Float16* __restrict__ Kd,
    const _Float16* __restrict__ Vd, _Float16* __restrict__ Od,
    const float* __restrict__ log_temp)
{
  __shared__ _Float16 sK[32*72];      // [j][d]  rows j, stride 72 halves
  __shared__ _Float16 sV[64*40];      // [d][j]  V transposed, stride 40
  __shared__ _Float16 sP[8*16*40];    // per-wave P staging, stride 40
  const int tid  = threadIdx.x, lane = tid & 31, wid = tid >> 5;
  const int lrow = lane & 15, lhi = lane >> 4;
  const int bh = blockIdx.y;
  const int b  = bh >> 3, h = bh & 7;
  const int q0 = blockIdx.x * 128;
  const long base = (long)bh * 2048;

  const float temp = __expf(log_temp[0]);

  // Preload this lane's Q fragments (d 0..31 and 32..63)
  const _Float16* qrow = Qd + (base + q0 + wid*16 + lrow)*64;
  v16h qf[2];
  qf[0] = load_frag_a(qrow,      lhi);
  qf[1] = load_frag_a(qrow + 32, lhi);

  const v8f z8 = {0.f,0.f,0.f,0.f,0.f,0.f,0.f,0.f};
  v8f oacc[4];
  float mi[8], li[8];
  #pragma unroll
  for (int t=0;t<4;++t) oacc[t]=z8;
  #pragma unroll
  for (int i=0;i<8;++i){ mi[i] = -3.0e38f; li[i] = 0.f; }

  const int rr = tid >> 3, cc = (tid & 7)*8;     // 32 rows x 8 chunks staging

  for (int j0 = 0; j0 < 2048; j0 += 32){
    __syncthreads();                              // prev compute done
#if HAVE_ASYNC_LDS
    ASYNC_COPY_B128(Kd + (base + j0 + rr)*64 + cc, sK + rr*72 + cc);
#else
    *(v8h*)(sK + rr*72 + cc) = *(const v8h*)(Kd + (base + j0 + rr)*64 + cc);
#endif
    {
      v8h tv = *(const v8h*)(Vd + (base + j0 + rr)*64 + cc);
      #pragma unroll
      for (int i=0;i<8;++i) sV[(cc+i)*40 + rr] = tv[i];   // transpose into LDS
    }
    WAIT_ASYNC();
    __syncthreads();

    // S = Q K^T  (two 16x16 column tiles, K-dim = d = 64 in 2 steps)
    v8f s0 = z8, s1 = z8;
    #pragma unroll
    for (int ks=0; ks<2; ++ks){
      v16h kf0 = load_frag_b(sK + lrow*72      + ks*32, lhi);
      v16h kf1 = load_frag_b(sK + (16+lrow)*72 + ks*32, lhi);
      s0 = WMMA_F32_F16(qf[ks], kf0, s0);
      s1 = WMMA_F32_F16(qf[ks], kf1, s1);
    }

    // scale, mask diagonal, online softmax (rows live on 16-lane half-groups)
    const int qg0 = q0 + wid*16 + lhi*8;
    float corr[8];
    #pragma unroll
    for (int i=0;i<8;++i){
      float a0 = s0[i]*temp, a1 = s1[i]*temp;
      if (qg0 + i == j0 + lrow)      a0 = -3.0e38f;
      if (qg0 + i == j0 + 16 + lrow) a1 = -3.0e38f;
      float mx = fmaxf(a0, a1);
      #pragma unroll
      for (int off=1; off<16; off<<=1) mx = fmaxf(mx, __shfl_xor(mx, off, 32));
      float mnew = fmaxf(mi[i], mx);
      float cr = __expf(mi[i] - mnew);
      float p0 = __expf(a0 - mnew), p1 = __expf(a1 - mnew);
      float rs = p0 + p1;
      #pragma unroll
      for (int off=1; off<16; off<<=1) rs += __shfl_xor(rs, off, 32);
      li[i] = li[i]*cr + rs;
      mi[i] = mnew;  corr[i] = cr;
      s0[i] = p0;    s1[i] = p1;
    }
    #pragma unroll
    for (int t=0;t<4;++t)
      #pragma unroll
      for (int i=0;i<8;++i) oacc[t][i] *= corr[i];

    // Repack P (C-layout f32 -> A-layout f16) via this wave's LDS region.
    // Same-wave LDS ops are in-order, so no barrier needed.
    _Float16* pw = sP + wid*(16*40);
    #pragma unroll
    for (int i=0;i<8;++i){
      pw[(lhi*8+i)*40 + lrow]      = (_Float16)s0[i];
      pw[(lhi*8+i)*40 + 16 + lrow] = (_Float16)s1[i];
    }
    v16h pf = load_frag_a(pw + lrow*40, lhi);

    // O += P V   (4 d-tiles)
    #pragma unroll
    for (int t=0;t<4;++t){
      v16h vf = load_frag_b(sV + (t*16+lrow)*40, lhi);
      oacc[t] = WMMA_F32_F16(pf, vf, oacc[t]);
    }
  }

  // Normalize and write f16 output in [b, n, h*64 + d] layout
  #pragma unroll
  for (int i=0;i<8;++i){
    float inv = 1.0f / li[i];
    long n = q0 + wid*16 + lhi*8 + i;
    _Float16* orow = Od + ((long)b*2048 + n)*512 + h*64;
    #pragma unroll
    for (int t=0;t<4;++t)
      orow[t*16 + lrow] = (_Float16)(oacc[t][i]*inv);
  }
}

// ---------------------------------------------------------------------------
// Small prep kernels: f32->f16 cast and f32->f16 transpose (weights)
// ---------------------------------------------------------------------------
__global__ void cast_f32_f16(const float* __restrict__ in,
                             _Float16* __restrict__ out, int n){
  int i = blockIdx.x*256 + threadIdx.x;
  if (i < n) out[i] = (_Float16)in[i];
}
__global__ void transpose_f32_f16(const float* __restrict__ in,
                                  _Float16* __restrict__ out, int R, int C){
  int i = blockIdx.x*256 + threadIdx.x;
  if (i < R*C){
    int r = i / C, c = i - r*C;
    out[(long)c*R + r] = (_Float16)in[i];
  }
}

// ---------------------------------------------------------------------------
extern "C" void kernel_launch(void* const* d_in, const int* in_sizes, int n_in,
                              void* d_out, int out_size, void* d_ws, size_t ws_size,
                              hipStream_t stream)
{
  (void)in_sizes; (void)n_in; (void)out_size; (void)ws_size;
  const float* x        = (const float*)d_in[0];   // [4,2048,512]
  const float* w_qkv    = (const float*)d_in[1];   // [512,1536]
  const float* w_out    = (const float*)d_in[2];   // [512,512]
  const float* b_out    = (const float*)d_in[3];   // [512]
  const float* log_temp = (const float*)d_in[4];   // scalar

  const long Mrows = 4L*2048;                       // 8192
  char* p = (char*)d_ws;
  _Float16* xh  = (_Float16*)p; p += Mrows*512*sizeof(_Float16);     // A for QKV
  _Float16* wqT = (_Float16*)p; p += 1536L*512*sizeof(_Float16);     // w_qkv^T
  _Float16* woT = (_Float16*)p; p += 512L*512*sizeof(_Float16);      // w_out^T
  _Float16* qh  = (_Float16*)p; p += 32L*2048*64*sizeof(_Float16);   // [b,h,n,d]
  _Float16* kh  = (_Float16*)p; p += 32L*2048*64*sizeof(_Float16);
  _Float16* vh  = (_Float16*)p; p += 32L*2048*64*sizeof(_Float16);
  _Float16* ao  = (_Float16*)p; p += Mrows*512*sizeof(_Float16);     // attn out

  cast_f32_f16<<<(int)((Mrows*512 + 255)/256), 256, 0, stream>>>(x, xh, (int)(Mrows*512));
  transpose_f32_f16<<<(512*1536 + 255)/256, 256, 0, stream>>>(w_qkv, wqT, 512, 1536);
  transpose_f32_f16<<<(512*512  + 255)/256, 256, 0, stream>>>(w_out, woT, 512, 512);

  // QKV projection: [8192,512] x [512,1536], scatter into per-head q/k/v
  gemm_wmma_f16<<<dim3((unsigned)(Mrows/128), 1536/128), 256, 0, stream>>>(
      xh, wqT, 512, 1536, 1, nullptr, nullptr, qh, kh, vh);

  // Attention: 16 query blocks x 32 (b,h) pairs
  attn_wmma<<<dim3(2048/128, 32), 256, 0, stream>>>(qh, kh, vh, ao, log_temp);

  // Output projection + bias: [8192,512] x [512,512] -> f32 d_out
  gemm_wmma_f16<<<dim3((unsigned)(Mrows/128), 512/128), 256, 0, stream>>>(
      ao, woT, 512, 512, 0, (float*)d_out, b_out, nullptr, nullptr, nullptr);
}
